// BasicBlock_29059748724989
// MI455X (gfx1250) — 8-bit matrix operand order repaired and checked by reading, compiled, not run on hardware
//
#include <hip/hip_runtime.h>


typedef int   v8i __attribute__((ext_vector_type(8)));
typedef float v8f __attribute__((ext_vector_type(8)));
typedef float v2f __attribute__((ext_vector_type(2)));
typedef int   vi4 __attribute__((vector_size(16)));
typedef __attribute__((address_space(1))) vi4* gvi4p;
typedef __attribute__((address_space(3))) vi4* lvi4p;

#define BN_EPS 1e-5f
#define NB   16
#define CIN  32
#define COUT 64
#define HIN  224
#define WIN  224
#define HOUT 112
#define WOUT 112
#define KPAD 320
#define PB_INTS (4*5*8*32)
#define PB_BYTES ((size_t)PB_INTS * 4)
#define SIGNS_BYTES ((size_t)NB * HIN * WIN * CIN)
#define POOL_BYTES  ((size_t)NB * HOUT * WOUT * CIN * 4)

#if __has_builtin(__builtin_amdgcn_global_load_async_to_lds_b128) && \
    __has_builtin(__builtin_amdgcn_s_wait_asynccnt)
#define USE_ASYNC_LDS 1
#else
#define USE_ASYNC_LDS 0
#endif

__global__ void pack_wb_kernel(const float* __restrict__ w_body,
                               int* __restrict__ pb) {
    int idx = blockIdx.x * blockDim.x + threadIdx.x;
    if (idx >= PB_INTS) return;
    int lane = idx & 31;
    int j    = (idx >> 5) & 7;
    int kc   = (idx >> 8) % 5;
    int nt   = idx / (5 * 8 * 32);
    int g    = lane >> 4;
    int n    = lane & 15;
    int co   = nt * 16 + n;
    int hf   = j >> 2, jj = j & 3;
    int kbase = kc * 64 + hf * 32 + g * 16 + jj * 4;
    unsigned int packed = 0u;
    #pragma unroll
    for (int b = 0; b < 4; ++b) {
        int kk = kbase + b;
        int t  = kk >> 5;
        int ci = kk & 31;
        int sb = 0;
        if (t < 9) {
            float w = w_body[co * (CIN * 9) + ci * 9 + t];
            sb = (w >= 0.0f) ? 1 : -1;
        }
        packed |= ((unsigned int)(sb & 0xFF)) << (8 * b);
    }
    pb[idx] = (int)packed;
}

__global__ __launch_bounds__(256) void prep_kernel(
    const float* __restrict__ x,
    signed char* __restrict__ signs,
    float* __restrict__ pooled) {
    __shared__ float sX[CIN * 2 * 32];

    const int tid = threadIdx.x, bx = blockIdx.x;
    const int wo_t = bx % 7;
    const int ho   = (bx / 7) % HOUT;
    const int b    = bx / (7 * HOUT);
    const int wo0  = wo_t * 16;
    const int wi0  = wo0 * 2;
    const int hi0  = ho * 2;
    const float* xb = x + (size_t)b * CIN * HIN * WIN;

    for (int e = tid; e < 512; e += 256) {
        int ci = e >> 4;
        int r  = (e >> 3) & 1;
        int q  = e & 7;
        const float4 v = *(const float4*)(xb + (size_t)ci * (HIN * WIN)
                                          + (hi0 + r) * WIN + wi0 + q * 4);
        *(float4*)&sX[(ci * 2 + r) * 32 + q * 4] = v;
    }
    __syncthreads();

    for (int e = tid; e < 512; e += 256) {
        int c4 = e & 7;
        int i  = e >> 3;
        int wl = i & 31;
        int r  = i >> 5;
        unsigned int packed = 0u;
        #pragma unroll
        for (int bb = 0; bb < 4; ++bb) {
            float v = sX[((c4 * 4 + bb) * 2 + r) * 32 + wl];
            packed |= ((unsigned int)((v >= 0.0f ? 1 : -1) & 0xFF)) << (8 * bb);
        }
        *(int*)&signs[(((size_t)(b * HIN) + hi0 + r) * WIN + wi0 + wl) * CIN
                      + c4 * 4] = (int)packed;
    }

    for (int e = tid; e < 512; e += 256) {
        int ci = e & 31;
        int wl = (e >> 5) * 2;
        float s = sX[(ci * 2 + 0) * 32 + wl] + sX[(ci * 2 + 0) * 32 + wl + 1]
                + sX[(ci * 2 + 1) * 32 + wl] + sX[(ci * 2 + 1) * 32 + wl + 1];
        pooled[(((size_t)(b * HOUT) + ho) * WOUT + wo0 + (e >> 5)) * CIN + ci]
            = 0.25f * s;
    }
}

__device__ __forceinline__ void tile_mma_store(
    const signed char* sA, const float (*sP)[16], float* sO,
    const int* __restrict__ pb, const float* __restrict__ w_ds,
    const float* __restrict__ g1, const float* __restrict__ b1,
    const float* __restrict__ m1, const float* __restrict__ v1,
    const float* __restrict__ g2, const float* __restrict__ b2,
    const float* __restrict__ m2, const float* __restrict__ v2,
    float* __restrict__ out, int b, int ho, int wo0, int tid) {

    const int lane = tid & 31;
    const int nt   = tid >> 5;
    const int g    = lane >> 4;
    const int ln   = lane & 15;

    v8i acc = {0, 0, 0, 0, 0, 0, 0, 0};
    #pragma unroll
    for (int kc = 0; kc < 5; ++kc) {
        v8i afrag, bfrag;
        #pragma unroll
        for (int j = 0; j < 8; ++j) {
            int kst = ((j >> 2) * 32) + ((j & 3) * 4) + g * 16;
            afrag[j] = *(const int*)&sA[ln * KPAD + kc * 64 + kst];
            bfrag[j] = pb[(((nt * 5 + kc) * 8) + j) * 32 + lane];
        }
        acc = __builtin_amdgcn_wmma_i32_16x16x64_iu8(
            true, afrag, true, bfrag, acc, false, false);
    }

    const int co = nt * 16 + ln;
    v8f racc = {0.f, 0.f, 0.f, 0.f, 0.f, 0.f, 0.f, 0.f};
#if __has_builtin(__builtin_amdgcn_wmma_f32_16x16x4_f32)
    #pragma unroll
    for (int kc = 0; kc < 8; ++kc) {
        v2f af = { sP[kc * 4 + g * 2 + 0][ln], sP[kc * 4 + g * 2 + 1][ln] };
        v2f bf = { w_ds[co * CIN + kc * 4 + g * 2 + 0],
                   w_ds[co * CIN + kc * 4 + g * 2 + 1] };
        racc = __builtin_amdgcn_wmma_f32_16x16x4_f32(
            false, af, false, bf, (short)0, racc, false, false);
    }
#else
    for (int ci = 0; ci < CIN; ++ci) {
        float w = w_ds[co * CIN + ci];
        #pragma unroll
        for (int r = 0; r < 8; ++r) racc[r] += w * sP[ci][r + 8 * g];
    }
#endif

    float inv1 = g1[co] * __frsqrt_rn(v1[co] + BN_EPS);
    float c1   = b1[co] - m1[co] * inv1;
    float inv2 = g2[co] * __frsqrt_rn(v2[co] + BN_EPS);
    float c2   = b2[co] - m2[co] * inv2;

    #pragma unroll
    for (int r = 0; r < 8; ++r) {
        sO[co * 16 + (r + 8 * g)] =
            (float)acc[r] * inv1 + c1 + racc[r] * inv2 + c2;
    }
    __syncthreads();

    float* ob = out + (((size_t)b * COUT) * HOUT + ho) * WOUT + wo0;
    for (int e = tid; e < COUT * 16; e += 128) {
        ob[(size_t)(e >> 4) * (HOUT * WOUT) + (e & 15)] = sO[e];
    }
}

__global__ __launch_bounds__(128) void fused_v2_kernel(
    const signed char* __restrict__ signs, const float* __restrict__ pooled,
    const int* __restrict__ pb, const float* __restrict__ w_ds,
    const float* __restrict__ g1, const float* __restrict__ b1,
    const float* __restrict__ m1, const float* __restrict__ v1,
    const float* __restrict__ g2, const float* __restrict__ b2,
    const float* __restrict__ m2, const float* __restrict__ v2,
    float* __restrict__ out) {

    __shared__ int   sAi[16 * KPAD / 4];
    __shared__ float sP[CIN][16];
    __shared__ float sO[COUT * 16];
    signed char* sA = (signed char*)sAi;

    const int tid = threadIdx.x, bx = blockIdx.x;
    const int wo_t = bx % 7;
    const int ho   = (bx / 7) % HOUT;
    const int b    = bx / (7 * HOUT);
    const int wo0  = wo_t * 16;

    for (int e = tid; e < 320; e += 128) {
        int m   = e / 20;
        int rem = e % 20;
        int t   = rem >> 1;
        int hf  = rem & 1;
        signed char* dst = sA + m * KPAD + t * 32 + hf * 16;
        const signed char* src = nullptr;
        if (t < 9) {
            int kh = t / 3, kw = t % 3;
            int hi = ho * 2 - 1 + kh;
            int wi = (wo0 + m) * 2 - 1 + kw;
            if (hi >= 0 && hi < HIN && wi >= 0 && wi < WIN)
                src = signs + (((size_t)(b * HIN) + hi) * WIN + wi) * CIN + hf * 16;
        }
#if USE_ASYNC_LDS
        if (src) {
            __builtin_amdgcn_global_load_async_to_lds_b128(
                (gvi4p)(void*)const_cast<signed char*>(src),
                (lvi4p)(void*)dst, 0, 0);
        } else {
            vi4 z = {0, 0, 0, 0};
            *(vi4*)dst = z;
        }
#else
        vi4 v = {0, 0, 0, 0};
        if (src) v = *(const vi4*)src;
        *(vi4*)dst = v;
#endif
    }

    for (int e = tid; e < CIN * 16; e += 128) {
        int ci = e & 31;
        int m  = e >> 5;
        sP[ci][m] = pooled[(((size_t)(b * HOUT) + ho) * WOUT + wo0 + m) * CIN + ci];
    }

#if USE_ASYNC_LDS
    __builtin_amdgcn_s_wait_asynccnt(0);
#endif
    __syncthreads();

    tile_mma_store(sA, sP, sO, pb, w_ds, g1, b1, m1, v1, g2, b2, m2, v2,
                   out, b, ho, wo0, tid);
}

__global__ __launch_bounds__(128) void fused_v1_kernel(
    const float* __restrict__ x,
    const int* __restrict__ pb, const float* __restrict__ w_ds,
    const float* __restrict__ g1, const float* __restrict__ b1,
    const float* __restrict__ m1, const float* __restrict__ v1,
    const float* __restrict__ g2, const float* __restrict__ b2,
    const float* __restrict__ m2, const float* __restrict__ v2,
    float* __restrict__ out) {

    __shared__ int   sAi[16 * KPAD / 4];
    __shared__ float sP[CIN][16];
    __shared__ float sO[COUT * 16];
    signed char* sA = (signed char*)sAi;

    const int tid = threadIdx.x, bx = blockIdx.x;
    const int wo_t = bx % 7;
    const int ho   = (bx / 7) % HOUT;
    const int b    = bx / (7 * HOUT);
    const int wo0  = wo_t * 16;
    const float* xb = x + (size_t)b * CIN * HIN * WIN;

    for (int e = tid; e < 16 * KPAD; e += 128) {
        int m  = e & 15;
        int k  = e >> 4;
        int t  = k >> 5;
        int ci = k & 31;
        signed char sv = 0;
        if (t < 9) {
            int kh = t / 3, kw = t % 3;
            int hi = ho * 2 - 1 + kh;
            int wi = (wo0 + m) * 2 - 1 + kw;
            if (hi >= 0 && hi < HIN && wi >= 0 && wi < WIN)
                sv = (xb[(size_t)ci * (HIN * WIN) + hi * WIN + wi] >= 0.0f) ? 1 : -1;
        }
        sA[m * KPAD + k] = sv;
    }
    for (int e = tid; e < CIN * 16; e += 128) {
        int m  = e & 15;
        int ci = e >> 4;
        const float* p = xb + (size_t)ci * (HIN * WIN)
                            + (ho * 2) * WIN + (wo0 + m) * 2;
        sP[ci][m] = 0.25f * (p[0] + p[1] + p[WIN] + p[WIN + 1]);
    }
    __syncthreads();

    tile_mma_store(sA, sP, sO, pb, w_ds, g1, b1, m1, v1, g2, b2, m2, v2,
                   out, b, ho, wo0, tid);
}

extern "C" void kernel_launch(void* const* d_in, const int* in_sizes, int n_in,
                              void* d_out, int out_size, void* d_ws, size_t ws_size,
                              hipStream_t stream) {
    const float* x      = (const float*)d_in[0];
    const float* w_body = (const float*)d_in[1];
    const float* g1     = (const float*)d_in[2];
    const float* b1     = (const float*)d_in[3];
    const float* m1     = (const float*)d_in[4];
    const float* v1     = (const float*)d_in[5];
    const float* w_ds   = (const float*)d_in[6];
    const float* g2     = (const float*)d_in[7];
    const float* b2     = (const float*)d_in[8];
    const float* m2     = (const float*)d_in[9];
    const float* v2     = (const float*)d_in[10];
    float* out = (float*)d_out;

    int* pb = (int*)d_ws;
    pack_wb_kernel<<<(PB_INTS + 255) / 256, 256, 0, stream>>>(w_body, pb);

    const int nblocks = NB * HOUT * (WOUT / 16);
    const size_t need = PB_BYTES + SIGNS_BYTES + POOL_BYTES;

    if (ws_size >= need) {
        signed char* signs = (signed char*)d_ws + PB_BYTES;
        float* pooled = (float*)((char*)d_ws + PB_BYTES + SIGNS_BYTES);
        prep_kernel<<<nblocks, 256, 0, stream>>>(x, signs, pooled);
        fused_v2_kernel<<<nblocks, 128, 0, stream>>>(
            signs, pooled, pb, w_ds, g1, b1, m1, v1, g2, b2, m2, v2, out);
    } else {
        fused_v1_kernel<<<nblocks, 128, 0, stream>>>(
            x, pb, w_ds, g1, b1, m1, v1, g2, b2, m2, v2, out);
    }
}
